// DynaFormer_49752901157521
// MI455X (gfx1250) — matrix operand order repaired and checked by reading, not compiled, not run on hardware
//
#include <hip/hip_runtime.h>
#include <hip/hip_bf16.h>

#define DEV __device__ __forceinline__

typedef __attribute__((ext_vector_type(16))) __bf16 v16bf;
typedef __attribute__((ext_vector_type(8)))  float  v8f;

namespace {
constexpr int N   = 2048;
constexpr int E   = 65536;
constexpr int INN = 16;
constexpr int D   = 128;
constexpr int H   = 8;
constexpr int FF  = 512;
constexpr int OUT = 64;
constexpr int HS  = 8;
constexpr int L   = 3;
constexpr int HD  = H * D;        // 1024
constexpr int HD3 = 3 * H * D;    // 3072
}

// ---------------- WMMA fragment helpers (per CDNA5 ISA layouts) ----------------

// A-matrix 16x32 bf16, row-major source with stride lda.
// lane<16: M=lane, K in {0..7, 16..23}; lane>=16: M=lane-16, K in {8..15, 24..31}
DEV v16bf load_a_frag(const __bf16* p, int lda) {
  int lane = threadIdx.x & 31;
  const __bf16* row = p + (lane & 15) * lda + ((lane >> 4) << 3);
  v16bf a;
#pragma unroll
  for (int i = 0; i < 8; ++i) a[i] = row[i];
#pragma unroll
  for (int i = 0; i < 8; ++i) a[8 + i] = row[16 + i];
  return a;
}

// B-matrix 32x16 bf16 where memory stores B^T, i.e. [Ncol][K] row-major, stride ld.
// lane&15 = N column; lane<16: K in {0..7, 16..23}; lane>=16: K in {8..15, 24..31}.
DEV v16bf load_bT_frag(const __bf16* p, int ld) {
  int lane = threadIdx.x & 31;
  const __bf16* q = p + (lane & 15) * ld + ((lane >> 4) << 3);
  v16bf b;
#pragma unroll
  for (int i = 0; i < 8; ++i) b[i] = q[i];
#pragma unroll
  for (int i = 0; i < 8; ++i) b[8 + i] = q[16 + i];
  return b;
}

DEV v8f wmma_bf16(v16bf a, v16bf b, v8f c) {
  return __builtin_amdgcn_wmma_f32_16x16x32_bf16(false, a, false, b, (short)0, c,
                                                 false, false);
}

// ---------------- gfx1250 async global->LDS helpers ----------------

DEV void async_b128(unsigned int lds, unsigned long long gaddr) {
  asm volatile("global_load_async_to_lds_b128 %0, %1, off"
               :: "v"(lds), "v"(gaddr)
               : "memory");
}

DEV unsigned int lds_off(const void* p) {
  // generic (flat) shared address: low 32 bits are the workgroup-relative LDS offset
  return (unsigned int)(unsigned long long)(uintptr_t)p;
}

// ---------------- Elementwise / setup kernels ----------------

__global__ __launch_bounds__(256) void zero_i32_kernel(int* p, int n) {
  int i = blockIdx.x * blockDim.x + threadIdx.x;
  if (i < n) p[i] = 0;
}

__global__ __launch_bounds__(256) void degree_kernel(const int* __restrict__ eidx,
                                                     int* __restrict__ outdeg,
                                                     int* __restrict__ indeg) {
  int i = blockIdx.x * blockDim.x + threadIdx.x;
  if (i < E) {
    atomicAdd(&outdeg[eidx[i]], 1);       // edge_index[0] = src -> out degree
    atomicAdd(&indeg[eidx[E + i]], 1);    // edge_index[1] = dst -> in degree
  }
}

__global__ __launch_bounds__(256) void embed_kernel(
    const float* __restrict__ x, const float* __restrict__ Wn,
    const float* __restrict__ bn, const float* __restrict__ z_in,
    const float* __restrict__ z_out, const int* __restrict__ indeg,
    const int* __restrict__ outdeg, float* __restrict__ h) {
  int idx = blockIdx.x * blockDim.x + threadIdx.x;
  if (idx >= N * D) return;
  int row = idx >> 7, d = idx & (D - 1);
  float s = bn[d];
  const float* xr = x + row * INN;
#pragma unroll
  for (int j = 0; j < INN; ++j) s += xr[j] * Wn[j * D + d];
  int di = min(indeg[row], 63);
  int dq = min(outdeg[row], 63);
  h[idx] = s + z_in[di * D + d] + z_out[dq * D + d];
}

__global__ __launch_bounds__(256) void spatial_bias_kernel(
    const float* __restrict__ pos, const float* __restrict__ mu,
    const float* __restrict__ sg, const float* __restrict__ wv,
    const float* __restrict__ b0, float* __restrict__ bias) {
  long idx = (long)blockIdx.x * blockDim.x + threadIdx.x;
  if (idx >= (long)N * N) return;
  int i = (int)(idx >> 11), j = (int)(idx & (N - 1));
  float dx = pos[i * 3 + 0] - pos[j * 3 + 0];
  float dy = pos[i * 3 + 1] - pos[j * 3 + 1];
  float dz = pos[i * 3 + 2] - pos[j * 3 + 2];
  float dist = sqrtf(dx * dx + dy * dy + dz * dz + 1e-12f);
  float acc = b0[0];
#pragma unroll
  for (int t = 0; t < HS; ++t) {
    float u = (dist - mu[t]) / sg[t];
    acc += wv[t] * __expf(-0.5f * u * u);
  }
  bias[idx] = acc;
}

// transpose fp32 [batch][R][C] -> bf16 [batch][C][R]
__global__ __launch_bounds__(256) void pack_wT_kernel(const float* __restrict__ src,
                                                      __bf16* __restrict__ dst,
                                                      int batch, int R, int C) {
  int idx = blockIdx.x * blockDim.x + threadIdx.x;
  int total = batch * R * C;
  if (idx >= total) return;
  int b = idx / (R * C);
  int rem = idx - b * (R * C);
  int r = rem / C, c = rem - r * C;
  dst[(long)b * R * C + (long)c * R + r] = (__bf16)src[idx];
}

__global__ __launch_bounds__(256) void pack_qkv_bias_kernel(
    const float* __restrict__ bq, const float* __restrict__ bk,
    const float* __restrict__ bv, float* __restrict__ dst) {
  int i = blockIdx.x * blockDim.x + threadIdx.x;
  if (i < HD) {
    dst[i] = bq[i];
    dst[HD + i] = bk[i];
    dst[2 * HD + i] = bv[i];
  }
}

// LayerNorm over D=128; one wave per row; bf16 output for WMMA consumers.
__global__ __launch_bounds__(128) void ln_kernel(const float* __restrict__ h,
                                                 const float* __restrict__ g,
                                                 const float* __restrict__ b,
                                                 __bf16* __restrict__ outp) {
  int row = (blockIdx.x * blockDim.x + threadIdx.x) >> 5;
  if (row >= N) return;
  int lane = threadIdx.x & 31;
  const float* p = h + (long)row * D;
  float v[4], s = 0.f;
#pragma unroll
  for (int i = 0; i < 4; ++i) {
    v[i] = p[lane + 32 * i];
    s += v[i];
  }
#pragma unroll
  for (int off = 16; off; off >>= 1) s += __shfl_xor(s, off, 32);
  float mean = s * (1.0f / D);
  float vs = 0.f;
#pragma unroll
  for (int i = 0; i < 4; ++i) {
    float dd = v[i] - mean;
    vs += dd * dd;
  }
#pragma unroll
  for (int off = 16; off; off >>= 1) vs += __shfl_xor(vs, off, 32);
  float rstd = rsqrtf(vs * (1.0f / D) + 1e-5f);
#pragma unroll
  for (int i = 0; i < 4; ++i) {
    int c = lane + 32 * i;
    outp[(long)row * D + c] = (__bf16)((v[i] - mean) * rstd * g[c] + b[c]);
  }
}

// ---------------- Generic bf16 WMMA GEMM: C = A[M,K] * Bt[N,K]^T + epilogue ----
// EPI: 0 = +bias ; 1 = +bias + resid ; 2 = +bias then exact GELU
// CbfT (optional): for cols >= 2*HD also write bf16 transposed (used to build Vt)
template <int EPI>
__global__ __launch_bounds__(128) void gemm_bf16_kernel(
    const __bf16* __restrict__ A, const __bf16* __restrict__ Bt,
    const float* __restrict__ bias, const float* __restrict__ resid,
    float* __restrict__ Cf, __bf16* __restrict__ Cbf, __bf16* __restrict__ CbfT,
    int M, int Nc, int K) {
  int wave = (blockIdx.x * blockDim.x + threadIdx.x) >> 5;
  int ntiles = Nc >> 4, mtiles = M >> 4;
  int tm = wave / ntiles, tn = wave - tm * ntiles;
  if (tm >= mtiles) return;
  const __bf16* a0 = A + (long)tm * 16 * K;
  const __bf16* b0 = Bt + (long)tn * 16 * K;
  v8f acc = {};
  for (int k = 0; k < K; k += 32) {
    v16bf a = load_a_frag(a0 + k, K);
    v16bf b = load_bT_frag(b0 + k, K);
    acc = wmma_bf16(a, b, acc);
  }
  int lane = threadIdx.x & 31;
  int col = tn * 16 + (lane & 15);
  int row0 = tm * 16 + ((lane >> 4) << 3);
  float bb = bias ? bias[col] : 0.f;
#pragma unroll
  for (int r = 0; r < 8; ++r) {
    float v = acc[r] + bb;
    long o = (long)(row0 + r) * Nc + col;
    if (EPI == 1) v += resid[o];
    if (EPI == 2) v = 0.5f * v * (1.0f + erff(v * 0.70710678118f));
    if (Cf) Cf[o] = v;
    if (Cbf) Cbf[o] = (__bf16)v;
    if (CbfT && col >= 2 * HD)
      CbfT[(long)(col - 2 * HD) * M + row0 + r] = (__bf16)v;
  }
}

// ---------------- Flash attention (cooperative, async-LDS staged) -------------
// Block = 4 waves, one head, 64 query rows (16 per wave). Double-buffered 32-key
// K/V tiles staged with global_load_async_to_lds_b128 (ASYNCcnt).
__global__ __launch_bounds__(128) void flash_attn_kernel(
    const __bf16* __restrict__ qkv,   // [N][3*H*D], cols: proj*HD + h*D + d
    const __bf16* __restrict__ vt,    // [H*D][N]  (V transposed)
    const float* __restrict__ bias,   // [N][N]
    __bf16* __restrict__ o) {         // [N][H*D]
  __shared__ __align__(16) __bf16 kbuf[2][32 * 128];   // [key][d]
  __shared__ __align__(16) __bf16 vbuf[2][128 * 32];   // [d][key]
  __shared__ __align__(16) __bf16 pstage[4][16 * 32];

  int tid = threadIdx.x;
  int widx = tid >> 5, lane = tid & 31;
  int hd = blockIdx.x >> 5;                    // 32 blocks per head
  int q0 = ((blockIdx.x & 31) << 6) + (widx << 4);

  const __bf16* qbase  = qkv + (long)q0 * HD3 + hd * D;
  const __bf16* kgbase = qkv + HD + hd * D;              // + key*HD3
  const __bf16* vgbase = vt + (long)hd * D * N;          // + d*N + key
  __bf16* stage = pstage[widx];

  v16bf qa[4];
#pragma unroll
  for (int kk = 0; kk < 4; ++kk) qa[kk] = load_a_frag(qbase + kk * 32, HD3);

  v8f oacc[8];
#pragma unroll
  for (int c = 0; c < 8; ++c) oacc[c] = (v8f){};
  float mrow[8], lrow[8];
#pragma unroll
  for (int r = 0; r < 8; ++r) { mrow[r] = -1e30f; lrow[r] = 0.f; }

  const float scale = 0.08838834764831845f;  // 1/sqrt(128)
  int ncol = lane & 15;
  int rbase = (lane >> 4) << 3;

  // cooperative tile prefetch: 8KB K tile + 8KB Vt tile, 16B per thread per slot
  auto prefetch = [&](int buf, int j) {
#pragma unroll
    for (int s = 0; s < 4; ++s) {
      unsigned int off = (unsigned int)((s * 128 + tid) * 16);
      unsigned int key = off >> 8, db = off & 255;       // K rows are 256B
      async_b128(lds_off(kbuf[buf]) + off,
                 (unsigned long long)(uintptr_t)(
                     (const char*)(kgbase + (long)(j + key) * HD3) + db));
    }
#pragma unroll
    for (int s = 0; s < 4; ++s) {
      unsigned int off = (unsigned int)((s * 128 + tid) * 16);
      unsigned int d = off >> 6, kb = off & 63;          // Vt rows are 64B
      async_b128(lds_off(vbuf[buf]) + off,
                 (unsigned long long)(uintptr_t)(
                     (const char*)(vgbase + (long)d * N + j) + kb));
    }
  };

  prefetch(0, 0);
  constexpr int NJT = N / 32;
  for (int jt = 0; jt < NJT; ++jt) {
    int buf = jt & 1;
    int j = jt * 32;
    if (jt + 1 < NJT) {
      prefetch(buf ^ 1, j + 32);
      asm volatile("s_wait_asynccnt 0x8" ::: "memory");  // current tile's 8 done
    } else {
      asm volatile("s_wait_asynccnt 0x0" ::: "memory");
    }
    __syncthreads();  // tile [buf] visible to all waves

    const __bf16* kl = kbuf[buf];
    v8f s0 = {}, s1 = {};
#pragma unroll
    for (int kk = 0; kk < 4; ++kk) {
      v16bf b0 = load_bT_frag(kl + kk * 32, 128);
      s0 = wmma_bf16(qa[kk], b0, s0);
      v16bf b1 = load_bT_frag(kl + 16 * 128 + kk * 32, 128);
      s1 = wmma_bf16(qa[kk], b1, s1);
    }
    float sv0[8], sv1[8], mnew[8];
#pragma unroll
    for (int r = 0; r < 8; ++r) {
      const float* brow = bias + (long)(q0 + rbase + r) * N + j;
      sv0[r] = s0[r] * scale + brow[ncol];
      sv1[r] = s1[r] * scale + brow[16 + ncol];
      mnew[r] = fmaxf(sv0[r], sv1[r]);
    }
#pragma unroll
    for (int off = 1; off < 16; off <<= 1)
#pragma unroll
      for (int r = 0; r < 8; ++r) mnew[r] = fmaxf(mnew[r], __shfl_xor(mnew[r], off, 32));
    float pscale[8], rsum[8];
#pragma unroll
    for (int r = 0; r < 8; ++r) {
      float m = fmaxf(mnew[r], mrow[r]);
      pscale[r] = __expf(mrow[r] - m);
      mrow[r] = m;
      sv0[r] = __expf(sv0[r] - m);
      sv1[r] = __expf(sv1[r] - m);
      rsum[r] = sv0[r] + sv1[r];
    }
#pragma unroll
    for (int off = 1; off < 16; off <<= 1)
#pragma unroll
      for (int r = 0; r < 8; ++r) rsum[r] += __shfl_xor(rsum[r], off, 32);
#pragma unroll
    for (int r = 0; r < 8; ++r) lrow[r] = lrow[r] * pscale[r] + rsum[r];
#pragma unroll
    for (int c = 0; c < 8; ++c)
#pragma unroll
      for (int r = 0; r < 8; ++r) oacc[c][r] *= pscale[r];
    // stage P (C-frag layout) -> reload as A fragment (wave-private LDS)
#pragma unroll
    for (int r = 0; r < 8; ++r) {
      stage[(rbase + r) * 32 + ncol] = (__bf16)sv0[r];
      stage[(rbase + r) * 32 + 16 + ncol] = (__bf16)sv1[r];
    }
    v16bf pa = load_a_frag(stage, 32);
    const __bf16* vl = vbuf[buf];
#pragma unroll
    for (int c = 0; c < 8; ++c) {
      v16bf vb = load_bT_frag(vl + (c * 16) * 32, 32);
      oacc[c] = wmma_bf16(pa, vb, oacc[c]);
    }
    __syncthreads();  // all waves done reading tile [buf] before it is reused
  }
  float inv[8];
#pragma unroll
  for (int r = 0; r < 8; ++r) inv[r] = 1.0f / lrow[r];
#pragma unroll
  for (int c = 0; c < 8; ++c)
#pragma unroll
    for (int r = 0; r < 8; ++r)
      o[(long)(q0 + rbase + r) * HD + hd * D + c * 16 + ncol] =
          (__bf16)(oacc[c][r] * inv[r]);
}

// ---------------- Host orchestration ----------------

static inline int gemm_blocks(int M, int Nc) {
  int tiles = (M >> 4) * (Nc >> 4);
  return (tiles + 3) / 4;  // 4 waves (128 threads) per block
}

extern "C" void kernel_launch(void* const* d_in, const int* in_sizes, int n_in,
                              void* d_out, int out_size, void* d_ws, size_t ws_size,
                              hipStream_t stream) {
  (void)in_sizes; (void)n_in; (void)out_size; (void)ws_size;
  const float* x     = (const float*)d_in[0];
  const int*   eidx  = (const int*)d_in[1];
  const float* pos   = (const float*)d_in[3];
  const float* nw    = (const float*)d_in[4];
  const float* nb    = (const float*)d_in[5];
  const float* z_in  = (const float*)d_in[8];
  const float* z_out = (const float*)d_in[9];
  const float* sp_mu = (const float*)d_in[10];
  const float* sp_sg = (const float*)d_in[11];
  const float* sp_w  = (const float*)d_in[12];
  const float* sp_b  = (const float*)d_in[13];
  const float* Wq = (const float*)d_in[14]; const float* bq = (const float*)d_in[15];
  const float* Wk = (const float*)d_in[16]; const float* bk = (const float*)d_in[17];
  const float* Wv = (const float*)d_in[18]; const float* bv = (const float*)d_in[19];
  const float* Wo = (const float*)d_in[20]; const float* bo = (const float*)d_in[21];
  const float* l1g = (const float*)d_in[22]; const float* l1b = (const float*)d_in[23];
  const float* l2g = (const float*)d_in[24]; const float* l2b = (const float*)d_in[25];
  const float* f1w = (const float*)d_in[26]; const float* f1b = (const float*)d_in[27];
  const float* f2w = (const float*)d_in[28]; const float* f2b = (const float*)d_in[29];
  const float* ow  = (const float*)d_in[30]; const float* ob  = (const float*)d_in[31];
  float* outp = (float*)d_out;

  char* wsp = (char*)d_ws;
  auto take = [&](size_t bytes) {
    char* p = wsp;
    wsp += (bytes + 255) & ~(size_t)255;
    return p;
  };
  int*    deg   = (int*)take((size_t)2 * N * 4);            // [outdeg | indeg]
  float*  h     = (float*)take((size_t)N * D * 4);
  float*  sbias = (float*)take((size_t)N * N * 4);
  __bf16* xn    = (__bf16*)take((size_t)N * D * 2);
  __bf16* qkvw  = (__bf16*)take((size_t)HD3 * D * 2);
  float*  qkvb  = (float*)take((size_t)HD3 * 4);
  __bf16* qkv   = (__bf16*)take((size_t)N * HD3 * 2);
  __bf16* vtb   = (__bf16*)take((size_t)HD * N * 2);
  __bf16* obuf  = (__bf16*)take((size_t)N * HD * 2);
  __bf16* wot   = (__bf16*)take((size_t)HD * D * 2);
  __bf16* f1t   = (__bf16*)take((size_t)FF * D * 2);
  __bf16* f2t   = (__bf16*)take((size_t)D * FF * 2);
  __bf16* hid   = (__bf16*)take((size_t)N * FF * 2);
  __bf16* owt   = (__bf16*)take((size_t)OUT * D * 2);
  __bf16* hbf   = (__bf16*)take((size_t)N * D * 2);
  int* outdeg = deg;
  int* indeg  = deg + N;

  // setup
  zero_i32_kernel<<<(2 * N + 255) / 256, 256, 0, stream>>>(deg, 2 * N);
  degree_kernel<<<(E + 255) / 256, 256, 0, stream>>>(eidx, outdeg, indeg);
  embed_kernel<<<(N * D + 255) / 256, 256, 0, stream>>>(x, nw, nb, z_in, z_out,
                                                        indeg, outdeg, h);
  spatial_bias_kernel<<<(int)(((long)N * N + 255) / 256), 256, 0, stream>>>(
      pos, sp_mu, sp_sg, sp_w, sp_b, sbias);
  pack_wT_kernel<<<(OUT * D + 255) / 256, 256, 0, stream>>>(ow, owt, 1, D, OUT);

  for (int l = 0; l < L; ++l) {
    // pack per-layer weights (transposed bf16)
    pack_wT_kernel<<<(H * D * D + 255) / 256, 256, 0, stream>>>(
        Wq + (long)l * H * D * D, qkvw + 0, H, D, D);
    pack_wT_kernel<<<(H * D * D + 255) / 256, 256, 0, stream>>>(
        Wk + (long)l * H * D * D, qkvw + (long)H * D * D, H, D, D);
    pack_wT_kernel<<<(H * D * D + 255) / 256, 256, 0, stream>>>(
        Wv + (long)l * H * D * D, qkvw + 2L * H * D * D, H, D, D);
    pack_qkv_bias_kernel<<<(HD + 255) / 256, 256, 0, stream>>>(
        bq + l * HD, bk + l * HD, bv + l * HD, qkvb);
    pack_wT_kernel<<<(HD * D + 255) / 256, 256, 0, stream>>>(
        Wo + (long)l * HD * D, wot, 1, HD, D);
    pack_wT_kernel<<<(D * FF + 255) / 256, 256, 0, stream>>>(
        f1w + (long)l * D * FF, f1t, 1, D, FF);
    pack_wT_kernel<<<(FF * D + 255) / 256, 256, 0, stream>>>(
        f2w + (long)l * FF * D, f2t, 1, FF, D);

    // attention block
    ln_kernel<<<(N * 32 + 127) / 128, 128, 0, stream>>>(h, l1g + l * D, l1b + l * D, xn);
    gemm_bf16_kernel<0><<<gemm_blocks(N, HD3), 128, 0, stream>>>(
        xn, qkvw, qkvb, nullptr, nullptr, qkv, vtb, N, HD3, D);
    flash_attn_kernel<<<H * (N / 64), 128, 0, stream>>>(qkv, vtb, sbias, obuf);
    gemm_bf16_kernel<1><<<gemm_blocks(N, D), 128, 0, stream>>>(
        obuf, wot, bo + l * D, h, h, nullptr, nullptr, N, D, HD);

    // FFN block
    ln_kernel<<<(N * 32 + 127) / 128, 128, 0, stream>>>(h, l2g + l * D, l2b + l * D, xn);
    gemm_bf16_kernel<2><<<gemm_blocks(N, FF), 128, 0, stream>>>(
        xn, f1t, f1b + l * FF, nullptr, nullptr, hid, nullptr, N, FF, D);
    gemm_bf16_kernel<1><<<gemm_blocks(N, D), 128, 0, stream>>>(
        hid, f2t, f2b + l * D, h, h, (l == L - 1) ? hbf : nullptr, nullptr, N, D, FF);
  }

  // output head
  gemm_bf16_kernel<0><<<gemm_blocks(N, OUT), 128, 0, stream>>>(
      hbf, owt, ob, nullptr, outp, nullptr, nullptr, N, OUT, D);
}
